// ContextBERTSelfAttention_2070174237062
// MI455X (gfx1250) — matrix operand order repaired and checked by reading, not compiled, not run on hardware
//
#include <hip/hip_runtime.h>
#include <math.h>

typedef __attribute__((ext_vector_type(16))) _Float16 v16h;
typedef __attribute__((ext_vector_type(8)))  float    v8f;

#define B_  4
#define S_  1024
#define D_  1024
#define H_  16
#define DH_ 64

// ---------------------------------------------------------------------------
// WMMA helpers (CDNA5 wave32, V_WMMA_F32_16X16X32_F16)
// ---------------------------------------------------------------------------
__device__ __forceinline__ v8f wmma_f16(v16h a, v16h b, v8f c) {
  // (neg_a, A, neg_b, B, c_mod, C, reuse_a, reuse_b)
  return __builtin_amdgcn_wmma_f32_16x16x32_f16(false, a, false, b, (short)0, c,
                                                false, false);
}

// A-matrix 16x32 (MxK), f16 row-major source with leading dim `ld`.
// Layout (matches B loaders): lanes 0-15 -> K in {0..15}; lanes 16-31 -> K in {16..31}
__device__ __forceinline__ v16h load_a_f16(const _Float16* p, int ld, int lane) {
  const int m  = lane & 15;
  const int kb = (lane < 16) ? 0 : 16;
  const _Float16* row = p + (size_t)m * ld + kb;
  v16h a;
#pragma unroll
  for (int j = 0; j < 8; ++j) a[j]     = row[j];
#pragma unroll
  for (int j = 0; j < 8; ++j) a[8 + j] = row[8 + j];
  return a;
}

// Same A layout but converting from f32 source on the fly.
__device__ __forceinline__ v16h load_a_f32(const float* p, int ld, int lane) {
  const int m  = lane & 15;
  const int kb = (lane < 16) ? 0 : 16;
  const float* row = p + (size_t)m * ld + kb;
  v16h a;
#pragma unroll
  for (int j = 0; j < 8; ++j) a[j]     = (_Float16)row[j];
#pragma unroll
  for (int j = 0; j < 8; ++j) a[8 + j] = (_Float16)row[8 + j];
  return a;
}

// B-matrix 32x16 (KxN) from a source where K is contiguous per N:
// element (k,n) at p[n*ld + k].  Lane = N, lanes 16-31 take K+16.
__device__ __forceinline__ v16h load_b_kcontig(const _Float16* p, int ld, int lane) {
  const int n  = lane & 15;
  const int kb = (lane < 16) ? 0 : 16;
  const _Float16* col = p + (size_t)n * ld + kb;
  v16h b;
#pragma unroll
  for (int j = 0; j < 16; ++j) b[j] = col[j];
  return b;
}

// B-matrix 32x16 from a row-major f32 weight matrix: element (k,n) at p[k*ld+n].
__device__ __forceinline__ v16h load_b_rowmajor_f32(const float* p, int ld, int lane) {
  const int n  = lane & 15;
  const int kb = (lane < 16) ? 0 : 16;
  v16h b;
#pragma unroll
  for (int j = 0; j < 16; ++j) b[j] = (_Float16)p[(size_t)(kb + j) * ld + n];
  return b;
}

__device__ __forceinline__ float sigmoidf_(float x) {
  return 1.0f / (1.0f + __expf(-x));
}

// ---------------------------------------------------------------------------
// 1) QKV projection: [4096,1024] x [1024,1024] + bias.
//    Block stages its 16x1024 activation row in LDS (f16, 32KB) once; 8 waves
//    sweep nt tiles for z in {q,k,v} from LDS.  q,k -> f16 [B,H,S,DH];
//    v -> transposed f16 [B,H,DH,S].
// ---------------------------------------------------------------------------
__global__ void __launch_bounds__(256) proj_qkv_kernel(
    const float* __restrict__ X,
    const float* __restrict__ Wq, const float* __restrict__ bq,
    const float* __restrict__ Wk, const float* __restrict__ bk,
    const float* __restrict__ Wv, const float* __restrict__ bv,
    _Float16* __restrict__ qh, _Float16* __restrict__ kh,
    _Float16* __restrict__ vt) {
  __shared__ _Float16 As[16 * D_];  // 32 KB
  const int lane = threadIdx.x & 31;
  const int wave = threadIdx.x >> 5;
  const int mt = blockIdx.x;  // 0..255 row tile over B*S

  // cooperative stage: 16x1024 f32 -> f16 in LDS (coalesced)
  const float* Xrow = X + (size_t)mt * 16 * D_;
#pragma unroll
  for (int i = 0; i < 64; ++i) {
    const int idx = threadIdx.x + i * 256;
    As[idx] = (_Float16)Xrow[idx];
  }
  __syncthreads();

  const float* Ws[3] = {Wq, Wk, Wv};
  const float* Bs[3] = {bq, bk, bv};
  const int mbase = mt * 16 + ((lane < 16) ? 0 : 8);

#pragma unroll
  for (int z = 0; z < 3; ++z) {
    const float* W = Ws[z];
#pragma unroll
    for (int i = 0; i < 2; ++i) {
      const int nt = blockIdx.y * 16 + i * 8 + wave;  // 0..63
      const float* B0 = W + nt * 16;
      v8f acc = {};
      for (int k0 = 0; k0 < D_; k0 += 32) {
        if (k0 + 32 < D_)
          __builtin_prefetch(B0 + (size_t)(k0 + 32) * D_, 0, 0);
        v16h a = load_a_f16(As + k0, D_, lane);
        v16h b = load_b_rowmajor_f32(B0 + (size_t)k0 * D_, D_, lane);
        acc = wmma_f16(a, b, acc);
      }
      const int n = nt * 16 + (lane & 15);
      const float bn = Bs[z][n];
      const int h = n >> 6, dh = n & 63;
#pragma unroll
      for (int r = 0; r < 8; ++r) {
        const int m = mbase + r;
        const int bb = m >> 10, s = m & 1023;
        const _Float16 val = (_Float16)(acc[r] + bn);
        if (z == 0)
          qh[((size_t)(bb * H_ + h) * S_ + s) * DH_ + dh] = val;
        else if (z == 1)
          kh[((size_t)(bb * H_ + h) * S_ + s) * DH_ + dh] = val;
        else
          vt[((size_t)(bb * H_ + h) * DH_ + dh) * S_ + s] = val;
      }
    }
  }
}

// ---------------------------------------------------------------------------
// 2) Context projections: [4096,1024] x [1024,64] + bias -> cq/ck f16 [B,S,DH]
//    Same LDS staging trick; 4 waves cover the 4 nt tiles, z in {cq,ck}.
// ---------------------------------------------------------------------------
__global__ void __launch_bounds__(128) proj_ctx_kernel(
    const float* __restrict__ Xc,
    const float* __restrict__ Wcq, const float* __restrict__ bcq,
    const float* __restrict__ Wck, const float* __restrict__ bck,
    _Float16* __restrict__ cqh, _Float16* __restrict__ ckh) {
  __shared__ _Float16 As[16 * D_];  // 32 KB
  const int lane = threadIdx.x & 31;
  const int wave = threadIdx.x >> 5;  // nt 0..3
  const int mt = blockIdx.x;          // 0..255

  const float* Xrow = Xc + (size_t)mt * 16 * D_;
#pragma unroll
  for (int i = 0; i < 128; ++i) {
    const int idx = threadIdx.x + i * 128;
    As[idx] = (_Float16)Xrow[idx];
  }
  __syncthreads();

  const float* Ws[2] = {Wcq, Wck};
  const float* Bs[2] = {bcq, bck};
  const int n = wave * 16 + (lane & 15);
  const int mbase = mt * 16 + ((lane < 16) ? 0 : 8);

#pragma unroll
  for (int z = 0; z < 2; ++z) {
    const float* B0 = Ws[z] + wave * 16;
    v8f acc = {};
    for (int k0 = 0; k0 < D_; k0 += 32) {
      v16h a = load_a_f16(As + k0, D_, lane);
      v16h b = load_b_rowmajor_f32(B0 + (size_t)k0 * DH_, DH_, lane);
      acc = wmma_f16(a, b, acc);
    }
    const float bn = Bs[z][n];
    _Float16* dst = z ? ckh : cqh;
#pragma unroll
    for (int r = 0; r < 8; ++r) {
      const int m = mbase + r;
      dst[(size_t)m * DH_ + n] = (_Float16)(acc[r] + bn);
    }
  }
}

// ---------------------------------------------------------------------------
// 3) lambda_context[b,h,s] = 1 - sigmoid(cq.wlqc + q.wlqq) - sigmoid(ck.wlkc + k.wlkk)
// ---------------------------------------------------------------------------
__global__ void __launch_bounds__(256) lambda_kernel(
    const _Float16* __restrict__ qh, const _Float16* __restrict__ kh,
    const _Float16* __restrict__ cqh, const _Float16* __restrict__ ckh,
    const float* __restrict__ w_lqc, const float* __restrict__ w_lqq,
    const float* __restrict__ w_lkc, const float* __restrict__ w_lkk,
    float* __restrict__ lam) {
  const int idx = blockIdx.x * 256 + threadIdx.x;  // (b*16+h)*1024+s
  const int s = idx & 1023;
  const int h = (idx >> 10) & 15;
  const int b = idx >> 14;
  const _Float16* cq = cqh + ((size_t)b * S_ + s) * DH_;
  const _Float16* ck = ckh + ((size_t)b * S_ + s) * DH_;
  const _Float16* qr = qh + ((size_t)(b * H_ + h) * S_ + s) * DH_;
  const _Float16* kr = kh + ((size_t)(b * H_ + h) * S_ + s) * DH_;
  float aq = 0.f, bq2 = 0.f, ak = 0.f, bk2 = 0.f;
#pragma unroll 8
  for (int d = 0; d < DH_; ++d) {
    aq  += (float)cq[d] * w_lqc[d];
    bq2 += (float)qr[d] * w_lqq[d];
    ak  += (float)ck[d] * w_lkc[d];
    bk2 += (float)kr[d] * w_lkk[d];
  }
  const float lq = sigmoidf_(aq + bq2);
  const float lk = sigmoidf_(ak + bk2);
  lam[idx] = 1.0f - (lq + lk);
}

// ---------------------------------------------------------------------------
// 4) Raw scores = q.k^T * scale + mask.  One wave owns a whole (b,h,qtile)
//    row: A fragments loaded once, 64 K-tiles swept (128 WMMAs / wave).
// ---------------------------------------------------------------------------
__global__ void __launch_bounds__(256) scores_kernel(
    const _Float16* __restrict__ qh, const _Float16* __restrict__ kh,
    const float* __restrict__ mask, float* __restrict__ probs) {
  const int lane = threadIdx.x & 31;
  const int wave = threadIdx.x >> 5;
  const int t = blockIdx.x * 8 + wave;  // 0..4095 : (b,h,qt)
  const int qt = t & 63, h = (t >> 6) & 15, b = t >> 10;
  const _Float16* A0 = qh + ((size_t)(b * H_ + h) * S_ + qt * 16) * DH_;
  const _Float16* Krow = kh + (size_t)(b * H_ + h) * S_ * DH_;
  const v16h a0 = load_a_f16(A0, DH_, lane);
  const v16h a1 = load_a_f16(A0 + 32, DH_, lane);
  const int nl = lane & 15;
  const int rofs = (lane < 16) ? 0 : 8;
  float* rowdst = probs + ((size_t)(b * H_ + h) * S_ + qt * 16) * S_;
  const float* mrow = mask + b * S_;
  for (int kt = 0; kt < 64; ++kt) {
    const _Float16* B0 = Krow + (size_t)kt * 16 * DH_;
    v8f acc = {};
    acc = wmma_f16(a0, load_b_kcontig(B0, DH_, lane), acc);
    acc = wmma_f16(a1, load_b_kcontig(B0 + 32, DH_, lane), acc);
    const int n = kt * 16 + nl;
    const float mk = mrow[n];
    float* dst = rowdst + n;
#pragma unroll
    for (int r = 0; r < 8; ++r)
      dst[(size_t)(r + rofs) * S_] = acc[r] * 0.125f + mk;
  }
}

// ---------------------------------------------------------------------------
// 5) quasi sigmoid scores = sigmoid(cq.ck^T*scale + mask), head-invariant,
//    stored into the h==0 slice of the quasi output region.
// ---------------------------------------------------------------------------
__global__ void __launch_bounds__(256) quasi_kernel(
    const _Float16* __restrict__ cqh, const _Float16* __restrict__ ckh,
    const float* __restrict__ mask, float* __restrict__ quasi) {
  const int lane = threadIdx.x & 31;
  const int wave = threadIdx.x >> 5;
  const int t = blockIdx.x * 8 + wave;  // 0..255 : (b,qt)
  const int qt = t & 63, b = t >> 6;
  const _Float16* A0 = cqh + ((size_t)b * S_ + qt * 16) * DH_;
  const _Float16* Krow = ckh + (size_t)b * S_ * DH_;
  const v16h a0 = load_a_f16(A0, DH_, lane);
  const v16h a1 = load_a_f16(A0 + 32, DH_, lane);
  const int nl = lane & 15;
  const int rofs = (lane < 16) ? 0 : 8;
  float* rowdst = quasi + ((size_t)(b * H_) * S_ + qt * 16) * S_;  // h==0 slice
  const float* mrow = mask + b * S_;
  for (int kt = 0; kt < 64; ++kt) {
    const _Float16* B0 = Krow + (size_t)kt * 16 * DH_;
    v8f acc = {};
    acc = wmma_f16(a0, load_b_kcontig(B0, DH_, lane), acc);
    acc = wmma_f16(a1, load_b_kcontig(B0 + 32, DH_, lane), acc);
    const int n = kt * 16 + nl;
    const float mk = mrow[n];
    float* dst = rowdst + n;
#pragma unroll
    for (int r = 0; r < 8; ++r)
      dst[(size_t)(r + rofs) * S_] = sigmoidf_(acc[r] * 0.125f + mk);
  }
}

// ---------------------------------------------------------------------------
// 6) Row-wise softmax in place (one 256-thread block per row of 1024)
// ---------------------------------------------------------------------------
__global__ void __launch_bounds__(256) softmax_kernel(float* __restrict__ probs) {
  __shared__ float red[256];
  const int t = threadIdx.x;
  float* p = probs + (size_t)blockIdx.x * S_;
  float x0 = p[t], x1 = p[t + 256], x2 = p[t + 512], x3 = p[t + 768];
  float m = fmaxf(fmaxf(x0, x1), fmaxf(x2, x3));
  red[t] = m;
  __syncthreads();
  for (int o = 128; o > 0; o >>= 1) {
    if (t < o) red[t] = fmaxf(red[t], red[t + o]);
    __syncthreads();
  }
  m = red[0];
  __syncthreads();
  x0 = __expf(x0 - m); x1 = __expf(x1 - m);
  x2 = __expf(x2 - m); x3 = __expf(x3 - m);
  red[t] = x0 + x1 + x2 + x3;
  __syncthreads();
  for (int o = 128; o > 0; o >>= 1) {
    if (t < o) red[t] += red[t + o];
    __syncthreads();
  }
  const float inv = 1.0f / red[0];
  p[t] = x0 * inv; p[t + 256] = x1 * inv;
  p[t + 512] = x2 * inv; p[t + 768] = x3 * inv;
}

// ---------------------------------------------------------------------------
// 7) quasi[b,h,q,k] = lam[b,h,q] * sig(b,q,k);  new = probs + quasi
//    sig read from h==0 slice before it is overwritten (same thread only).
// ---------------------------------------------------------------------------
__global__ void __launch_bounds__(256) combine_kernel(
    const float* __restrict__ lam, const float* __restrict__ probs,
    float* __restrict__ newp, float* __restrict__ quasi) {
  const size_t idx = (size_t)blockIdx.x * 256 + threadIdx.x;  // (b,q,k)
  const int k = (int)(idx & 1023);
  const int q = (int)((idx >> 10) & 1023);
  const int b = (int)(idx >> 20);
  const size_t base0 = (((size_t)b * H_ * S_) + q) * S_ + k;
  const float sig = quasi[base0];
#pragma unroll
  for (int h = 0; h < H_; ++h) {
    const size_t o = base0 + (size_t)h * S_ * S_;
    const float qa = lam[(b * H_ + h) * S_ + q] * sig;
    quasi[o] = qa;
    newp[o]  = probs[o] + qa;
  }
}

// ---------------------------------------------------------------------------
// 8) ctx = new_probs @ v.  One wave owns (b,h,qtile) and all 4 DH tiles:
//    each expensive A fragment (f32 probs -> f16) is reused by 4 WMMAs.
// ---------------------------------------------------------------------------
__global__ void __launch_bounds__(256) av_kernel(
    const float* __restrict__ newp, const _Float16* __restrict__ vt,
    float* __restrict__ ctx) {
  const int lane = threadIdx.x & 31;
  const int wave = threadIdx.x >> 5;
  const int t = blockIdx.x * 8 + wave;  // 0..4095 : (b,h,qt)
  const int qt = t & 63, h = (t >> 6) & 15, b = t >> 10;
  const float* A0 = newp + ((size_t)(b * H_ + h) * S_ + qt * 16) * S_;
  const _Float16* Vb = vt + (size_t)(b * H_ + h) * DH_ * S_;
  v8f acc[4] = {{}, {}, {}, {}};
  for (int k0 = 0; k0 < S_; k0 += 32) {
    if (k0 + 32 < S_) __builtin_prefetch(A0 + k0 + 32, 0, 0);
    const v16h a = load_a_f32(A0 + k0, S_, lane);
#pragma unroll
    for (int nt = 0; nt < 4; ++nt)
      acc[nt] = wmma_f16(a, load_b_kcontig(Vb + (size_t)nt * 16 * S_ + k0, S_, lane),
                         acc[nt]);
  }
  const int nl = lane & 15;
  const int rofs = (lane < 16) ? 0 : 8;
#pragma unroll
  for (int nt = 0; nt < 4; ++nt) {
    const int dh = nt * 16 + nl;
#pragma unroll
    for (int r = 0; r < 8; ++r) {
      const int s = qt * 16 + r + rofs;
      ctx[((size_t)b * S_ + s) * D_ + h * DH_ + dh] = acc[nt][r];
    }
  }
}

// ---------------------------------------------------------------------------
extern "C" void kernel_launch(void* const* d_in, const int* in_sizes, int n_in,
                              void* d_out, int out_size, void* d_ws, size_t ws_size,
                              hipStream_t stream) {
  const float* hidden = (const float*)d_in[0];
  const float* mask   = (const float*)d_in[1];
  const float* ctxemb = (const float*)d_in[2];
  const float* Wq  = (const float*)d_in[3];  const float* bq  = (const float*)d_in[4];
  const float* Wk  = (const float*)d_in[5];  const float* bk  = (const float*)d_in[6];
  const float* Wv  = (const float*)d_in[7];  const float* bv  = (const float*)d_in[8];
  const float* Wcq = (const float*)d_in[9];  const float* bcq = (const float*)d_in[10];
  const float* Wck = (const float*)d_in[11]; const float* bck = (const float*)d_in[12];
  const float* w_lqc = (const float*)d_in[13];
  const float* w_lqq = (const float*)d_in[14];
  const float* w_lkc = (const float*)d_in[15];
  const float* w_lkk = (const float*)d_in[16];

  // workspace: q/k f16 [B,H,S,DH], v^T f16 [B,H,DH,S], cq/ck f16 [B,S,DH], lam f32
  _Float16* qh  = (_Float16*)d_ws;
  _Float16* kh  = qh  + (size_t)B_ * H_ * S_ * DH_;
  _Float16* vt  = kh  + (size_t)B_ * H_ * S_ * DH_;
  _Float16* cqh = vt  + (size_t)B_ * H_ * S_ * DH_;
  _Float16* ckh = cqh + (size_t)B_ * S_ * DH_;
  float*    lam = (float*)(ckh + (size_t)B_ * S_ * DH_);

  // outputs, concatenated: ctx, new_probs, attention_probs, quasi_probs
  float* out   = (float*)d_out;
  float* ctx   = out;
  float* newp  = ctx  + (size_t)B_ * S_ * D_;
  float* probs = newp + (size_t)B_ * H_ * S_ * S_;
  float* quasi = probs + (size_t)B_ * H_ * S_ * S_;

  proj_qkv_kernel<<<dim3(256, 4), 256, 0, stream>>>(hidden, Wq, bq, Wk, bk,
                                                    Wv, bv, qh, kh, vt);
  proj_ctx_kernel<<<dim3(256, 1), 128, 0, stream>>>(ctxemb, Wcq, bcq, Wck, bck,
                                                    cqh, ckh);
  lambda_kernel<<<256, 256, 0, stream>>>(qh, kh, cqh, ckh, w_lqc, w_lqq, w_lkc,
                                         w_lkk, lam);
  scores_kernel<<<512, 256, 0, stream>>>(qh, kh, mask, probs);
  quasi_kernel<<<32, 256, 0, stream>>>(cqh, ckh, mask, quasi);
  softmax_kernel<<<65536, 256, 0, stream>>>(probs);
  combine_kernel<<<16384, 256, 0, stream>>>(lam, probs, newp, quasi);
  av_kernel<<<512, 256, 0, stream>>>(newp, vt, ctx);
}
